// MultiheadAttention_40166534152514
// MI455X (gfx1250) — matrix operand order repaired and checked by reading, not compiled, not run on hardware
//
#include <hip/hip_runtime.h>
#include <math.h>

typedef __attribute__((ext_vector_type(16))) _Float16 v16h;
typedef __attribute__((ext_vector_type(8)))  _Float16 v8h;
typedef __attribute__((ext_vector_type(8)))  float    v8f;
typedef __attribute__((ext_vector_type(4)))  float    v4f;

#define EDIM  1024
#define HEADS 16
#define SEQ   1024
#define BATCH 4
#define HDIM  64
#define NTOK  (SEQ*BATCH)   // 4096 tokens

#if __has_builtin(__builtin_amdgcn_global_load_async_to_lds_b128)
#define HAVE_ASYNC_LDS 1
#else
#define HAVE_ASYNC_LDS 0
#endif

// Builtin wants int4* in global (AS1) / LDS (AS3) address spaces.
typedef int v4i __attribute__((vector_size(16)));
typedef __attribute__((address_space(1))) v4i g_v4i;
typedef __attribute__((address_space(3))) v4i l_v4i;

static __device__ __forceinline__ g_v4i* gbl_cast4(const void* p) {
  // flat global VA == AS1 VA
  return (g_v4i*)(uintptr_t)p;
}
static __device__ __forceinline__ l_v4i* lds_cast4(void* p) {
  // low 32 bits of a flat LDS pointer are the LDS byte offset
  return (l_v4i*)(uint32_t)(uintptr_t)p;
}
static __device__ __forceinline__ void wait_async0() {
#if __has_builtin(__builtin_amdgcn_s_wait_asynccnt)
  __builtin_amdgcn_s_wait_asynccnt(0);
#else
  asm volatile("s_wait_asynccnt 0x0" ::: "memory");
#endif
}

// ---------------------------------------------------------------- helpers

static __device__ __forceinline__ v16h combine16(v8h lo, v8h hi) {
  return __builtin_shufflevector(lo, hi, 0,1,2,3,4,5,6,7,8,9,10,11,12,13,14,15);
}
// 16 contiguous halfs (B-fragment: K contiguous per lane, lanes = columns)
static __device__ __forceinline__ v16h ld_b16(const _Float16* p) {
  v8h lo = *(const v8h*)p;
  v8h hi = *(const v8h*)(p + 8);
  return combine16(lo, hi);
}
// A-fragment: lane half hl reads K chunk {c0..c0+15}, p = base + hl*16
static __device__ __forceinline__ v16h ld_a16(const _Float16* p) {
  v8h lo = *(const v8h*)p;
  v8h hi = *(const v8h*)(p + 8);
  return combine16(lo, hi);
}
static __device__ __forceinline__ v8f wmma_f16(v16h a, v16h b, v8f c) {
  return __builtin_amdgcn_wmma_f32_16x16x32_f16(false, a, false, b,
                                                (short)0, c, false, false);
}

// DPP16-based 16-lane xor reductions (pure VALU; co-executes with WMMA)
template<int CTRL>
static __device__ __forceinline__ float dppf(float x) {
  int r = __builtin_amdgcn_update_dpp(0, __float_as_int(x), CTRL, 0xF, 0xF, true);
  return __int_as_float(r);
}
static __device__ __forceinline__ float hmax16(float v) {
  v = fmaxf(v, dppf<0xB1>(v));    // quad_perm(1,0,3,2)  == xor 1
  v = fmaxf(v, dppf<0x4E>(v));    // quad_perm(2,3,0,1)  == xor 2
  v = fmaxf(v, dppf<0x141>(v));   // row_half_mirror     == xor 4 (post quad-reduce)
  v = fmaxf(v, dppf<0x140>(v));   // row_mirror          == xor 8 (post half-reduce)
  return v;
}
static __device__ __forceinline__ float hsum16(float v) {
  v += dppf<0xB1>(v);
  v += dppf<0x4E>(v);
  v += dppf<0x141>(v);
  v += dppf<0x140>(v);
  return v;
}

// ------------------------------------------------ f32 -> f16 convert (one-shot)
__global__ __launch_bounds__(256) void cvt_f16_kernel(
    const float* __restrict__ src, _Float16* __restrict__ dst, int n) {
  int i = (blockIdx.x * 256 + threadIdx.x) * 4;
  if (i < n) {
    v4f a = *(const v4f*)(src + i);
    dst[i + 0] = (_Float16)a.x; dst[i + 1] = (_Float16)a.y;
    dst[i + 2] = (_Float16)a.z; dst[i + 3] = (_Float16)a.w;
  }
}

// ------------------------------------------------ projection GEMM (x @ W^T + b)
// Y[m,n] = (sum_k A[m,k] * W[n,k] + bias[n]) * scale    (A, W already f16)
// Block: 256 thr = 8 waves (4m x 2n); block tile 128x128, BK=32; wave tile 32x64.
// Double-buffered LDS staging: async DMA of tile k+1 overlaps WMMA on tile k.
template<bool OUT_F16>
__global__ __launch_bounds__(256) void proj_gemm_kernel(
    const _Float16* __restrict__ A, const _Float16* __restrict__ Wh,
    const float* __restrict__ bias, float scale,
    _Float16* __restrict__ Y16, float* __restrict__ Y32) {
  constexpr int BM = 128, BN = 128, BK = 32, LDT = 56; // 112B row stride (16B mult)
  constexpr int N = EDIM, K = EDIM;
  __shared__ _Float16 As[2][BM * LDT];
  __shared__ _Float16 Bs[2][BN * LDT];

  const int tid  = threadIdx.x;
  const int lane = tid & 31;
  const int wave = tid >> 5;
  const int wm = wave >> 1, wn = wave & 1;
  const int m0 = blockIdx.x * BM, n0 = blockIdx.y * BN;
  const int hl = lane >> 4, l16 = lane & 15;

  const v8f vzero = {0.f,0.f,0.f,0.f,0.f,0.f,0.f,0.f};
  v8f acc[2][4];
  #pragma unroll
  for (int i = 0; i < 2; i++)
    #pragma unroll
    for (int j = 0; j < 4; j++) acc[i][j] = vzero;

  auto stage_tile = [&](int k0, int buf) {
    #pragma unroll
    for (int it = 0; it < 2; it++) {
      int idx = tid * 8 + it * 2048;
      int row = idx >> 5, col = idx & 31;
#if HAVE_ASYNC_LDS
      __builtin_amdgcn_global_load_async_to_lds_b128(
          gbl_cast4(A + (size_t)(m0 + row) * K + k0 + col),
          lds_cast4(&As[buf][row * LDT + col]), 0, 0);
      __builtin_amdgcn_global_load_async_to_lds_b128(
          gbl_cast4(Wh + (size_t)(n0 + row) * K + k0 + col),
          lds_cast4(&Bs[buf][row * LDT + col]), 0, 0);
#else
      *(v8h*)&As[buf][row * LDT + col] =
          *(const v8h*)(A + (size_t)(m0 + row) * K + k0 + col);
      *(v8h*)&Bs[buf][row * LDT + col] =
          *(const v8h*)(Wh + (size_t)(n0 + row) * K + k0 + col);
#endif
    }
  };

  stage_tile(0, 0);
  wait_async0();
  __syncthreads();

  #pragma unroll 1
  for (int k0 = 0; k0 < K; k0 += BK) {
    const int cur = (k0 >> 5) & 1;
    // kick off DMA of the next tile into the other buffer (overlaps with WMMA)
    if (k0 + BK < K) stage_tile(k0 + BK, cur ^ 1);
    // prefetch the tile after next into L2 (global_prefetch_b8)
    if (k0 + 2 * BK < K) {
      int row = tid >> 2, col = (tid & 3) * 8;
      __builtin_prefetch(A  + (size_t)(m0 + row) * K + k0 + 2 * BK + col, 0, 3);
      __builtin_prefetch(Wh + (size_t)(n0 + row) * K + k0 + 2 * BK + col, 0, 3);
    }

    v16h afrag[2], bfrag[4];
    #pragma unroll
    for (int i = 0; i < 2; i++) {
      int row = wm * 32 + i * 16 + l16;
      afrag[i] = ld_a16(&As[cur][row * LDT + hl * 16]);
    }
    #pragma unroll
    for (int j = 0; j < 4; j++) {
      int col = wn * 64 + j * 16 + l16;
      bfrag[j] = ld_b16(&Bs[cur][col * LDT + hl * 16]);
    }
    #pragma unroll
    for (int i = 0; i < 2; i++)
      #pragma unroll
      for (int j = 0; j < 4; j++)
        acc[i][j] = wmma_f16(afrag[i], bfrag[j], acc[i][j]);

    // next tile's DMA must be complete (and all waves past this tile's reads)
    wait_async0();
    __syncthreads();
  }

  // ---- epilogue: C layout (VGPR r -> row r+8*hl, lane -> column) -> memory
  #pragma unroll
  for (int j = 0; j < 4; j++) {
    int col = n0 + wn * 64 + j * 16 + l16;
    float bj = bias[col];
    #pragma unroll
    for (int i = 0; i < 2; i++) {
      int mrow = m0 + wm * 32 + i * 16 + hl * 8;
      #pragma unroll
      for (int r = 0; r < 8; r++) {
        float v = (acc[i][j][r] + bj) * scale;
        size_t off = (size_t)(mrow + r) * N + col;
        if (OUT_F16) Y16[off] = (_Float16)v;
        else         Y32[off] = v;
      }
    }
  }
}

// ------------------------------------------------ V transpose: vf(S,B,E) -> vt[bh][d][s]
__global__ __launch_bounds__(256) void transpose_v_kernel(
    const _Float16* __restrict__ vf, _Float16* __restrict__ vt) {
  __shared__ _Float16 tile[16][17];
  const int s0 = blockIdx.x * 16, d0 = blockIdx.y * 16, bh = blockIdx.z;
  const int b = bh >> 4, h = bh & 15;
  const int di = threadIdx.x & 15, si = threadIdx.x >> 4;
  tile[si][di] = vf[((size_t)(s0 + si) * BATCH + b) * EDIM + h * HDIM + d0 + di];
  __syncthreads();
  const int so = threadIdx.x & 15, dd = threadIdx.x >> 4;
  vt[((size_t)bh * HDIM + d0 + dd) * SEQ + s0 + so] = tile[so][dd];
}

// ------------------------------------------------ fused flash attention
// One wave handles 16 query rows of one (b,h). blockIdx = (bh, t-block of 128).
__global__ __launch_bounds__(256) void flash_attn_kernel(
    const _Float16* __restrict__ qf, const _Float16* __restrict__ kf,
    const _Float16* __restrict__ vt, const float* __restrict__ rbias,
    _Float16* __restrict__ out) {
  __shared__ _Float16 Pl[8][16 * 48];   // per-wave P relayout scratch (96B row stride)
  const int tid = threadIdx.x, lane = tid & 31, wave = tid >> 5;
  const int hl = lane >> 4, l16 = lane & 15;
  const int bh = blockIdx.x, b = bh >> 4, h = bh & 15;
  const int t0 = blockIdx.y * 128 + wave * 16;

  // Q fragments (A layout), D=64 => two K-steps of 32
  v16h qa0, qa1;
  {
    const _Float16* p = qf + ((size_t)(t0 + l16) * BATCH + b) * EDIM + h * HDIM + hl * 16;
    qa0 = ld_a16(p);
    qa1 = ld_a16(p + 32);
  }

  const v8f vzero = {0.f,0.f,0.f,0.f,0.f,0.f,0.f,0.f};
  v8f acc[4] = {vzero, vzero, vzero, vzero};
  float mrow[8], lrow[8];
  #pragma unroll
  for (int r = 0; r < 8; r++) { mrow[r] = -INFINITY; lrow[r] = 0.f; }

  const _Float16* vt_bh = vt + (size_t)bh * HDIM * SEQ;
  _Float16* pw = &Pl[wave][0];

  #pragma unroll 1
  for (int s0 = 0; s0 < SEQ; s0 += 32) {
    // ---- scores: Q (16x64) x K^T (64x32) -> two 16x16 C tiles
    const _Float16* kp0 = kf + ((size_t)(s0 + l16) * BATCH + b) * EDIM + h * HDIM + hl * 16;
    const _Float16* kp1 = kp0 + (size_t)16 * BATCH * EDIM;
    v16h kb00 = ld_b16(kp0),  kb01 = ld_b16(kp0 + 32);
    v16h kb10 = ld_b16(kp1),  kb11 = ld_b16(kp1 + 32);
    if (s0 + 32 < SEQ) {
      __builtin_prefetch(kp0 + (size_t)32 * BATCH * EDIM, 0, 3);
      __builtin_prefetch(vt_bh + (size_t)l16 * SEQ + s0 + 32, 0, 3);
    }
    v8f c0 = vzero, c1 = vzero;
    c0 = wmma_f16(qa0, kb00, c0);  c0 = wmma_f16(qa1, kb01, c0);
    c1 = wmma_f16(qa0, kb10, c1);  c1 = wmma_f16(qa1, kb11, c1);

    // ---- residual attention bias (pre-softmax)
    const float* bp = rbias + ((size_t)h * SEQ + (t0 + 8 * hl)) * SEQ + s0 + l16;
    #pragma unroll
    for (int r = 0; r < 8; r++) {
      c0[r] += bp[(size_t)r * SEQ];
      c1[r] += bp[(size_t)r * SEQ + 16];
    }

    // ---- online softmax (row stats via DPP16 half-wave reductions)
    float al[8];
    #pragma unroll
    for (int r = 0; r < 8; r++) {
      float tm = hmax16(fmaxf(c0[r], c1[r]));
      float mn = fmaxf(mrow[r], tm);
      al[r] = __expf(mrow[r] - mn);
      mrow[r] = mn;
      c0[r] = __expf(c0[r] - mn);
      c1[r] = __expf(c1[r] - mn);
      float ts = hsum16(c0[r] + c1[r]);
      lrow[r] = lrow[r] * al[r] + ts;
    }
    #pragma unroll
    for (int dt = 0; dt < 4; dt++)
      #pragma unroll
      for (int r = 0; r < 8; r++) acc[dt][r] *= al[r];

    // ---- relayout P: C layout -> LDS row-major -> A fragment
    #pragma unroll
    for (int r = 0; r < 8; r++) {
      int row = r + 8 * hl;
      pw[row * 48 + l16]      = (_Float16)c0[r];
      pw[row * 48 + 16 + l16] = (_Float16)c1[r];
    }
    // DS ops are in-order within a wave; compiler inserts s_wait_dscnt for the RAW.
    v16h pa = ld_a16(&pw[l16 * 48 + hl * 16]);

    // ---- O += P (16x32) x V (32x64): 4 d-tiles, V^T fragments contiguous
    #pragma unroll
    for (int dt = 0; dt < 4; dt++) {
      const _Float16* vp = vt_bh + (size_t)(dt * 16 + l16) * SEQ + s0 + hl * 16;
      v16h vb = ld_b16(vp);
      acc[dt] = wmma_f16(pa, vb, acc[dt]);
    }
  }

  // ---- normalize and write attn (T,B,E) as f16 for the O-projection
  #pragma unroll
  for (int r = 0; r < 8; r++) {
    float inv = 1.0f / lrow[r];
    int t = t0 + r + 8 * hl;
    size_t ob = ((size_t)t * BATCH + b) * EDIM + h * HDIM;
    #pragma unroll
    for (int dt = 0; dt < 4; dt++)
      out[ob + dt * 16 + l16] = (_Float16)(acc[dt][r] * inv);
  }
}

// ------------------------------------------------ launch
extern "C" void kernel_launch(void* const* d_in, const int* in_sizes, int n_in,
                              void* d_out, int out_size, void* d_ws, size_t ws_size,
                              hipStream_t stream) {
  (void)in_sizes; (void)n_in; (void)out_size; (void)ws_size;
  const float* query = (const float*)d_in[0];
  const float* key   = (const float*)d_in[1];
  const float* value = (const float*)d_in[2];
  const float* rbias = (const float*)d_in[3];
  const float* Wq = (const float*)d_in[4];  const float* bq = (const float*)d_in[5];
  const float* Wk = (const float*)d_in[6];  const float* bk = (const float*)d_in[7];
  const float* Wv = (const float*)d_in[8];  const float* bv = (const float*)d_in[9];
  const float* Wo = (const float*)d_in[10]; const float* bo = (const float*)d_in[11];

  char* ws = (char*)d_ws;
  const size_t MB = (size_t)1 << 20;
  _Float16* xq     = (_Float16*)(ws +  0 * MB);   // query f16  (dead after Q gemm)
  _Float16* xk     = (_Float16*)(ws +  8 * MB);   // key f16    (dead after K gemm)
  _Float16* xv     = (_Float16*)(ws + 16 * MB);   // value f16  (dead after V gemm)
  _Float16* wqh    = (_Float16*)(ws + 24 * MB);
  _Float16* wkh    = (_Float16*)(ws + 26 * MB);
  _Float16* wvh    = (_Float16*)(ws + 28 * MB);
  _Float16* woh    = (_Float16*)(ws + 30 * MB);
  _Float16* qf     = (_Float16*)(ws + 32 * MB);   // (T*B,E) f16, pre-scaled by D^-0.5
  _Float16* kf     = (_Float16*)(ws + 40 * MB);
  _Float16* vf     = (_Float16*)(ws + 48 * MB);
  _Float16* vt     = (_Float16*)(ws +  0 * MB);   // reuse xq: (BH,D,S) f16
  _Float16* attn16 = (_Float16*)(ws +  8 * MB);   // reuse xk: (T*B,E) f16
  // total workspace: 56 MB

  const int NA = NTOK * EDIM;   // 4M activations
  const int NW = EDIM * EDIM;   // 1M weights
  cvt_f16_kernel<<<NA / 1024, 256, 0, stream>>>(query, xq, NA);
  cvt_f16_kernel<<<NA / 1024, 256, 0, stream>>>(key,   xk, NA);
  cvt_f16_kernel<<<NA / 1024, 256, 0, stream>>>(value, xv, NA);
  cvt_f16_kernel<<<NW / 1024, 256, 0, stream>>>(Wq, wqh, NW);
  cvt_f16_kernel<<<NW / 1024, 256, 0, stream>>>(Wk, wkh, NW);
  cvt_f16_kernel<<<NW / 1024, 256, 0, stream>>>(Wv, wvh, NW);
  cvt_f16_kernel<<<NW / 1024, 256, 0, stream>>>(Wo, woh, NW);

  dim3 gGemm(NTOK / 128, EDIM / 128);            // 32 x 8
  proj_gemm_kernel<true ><<<gGemm, 256, 0, stream>>>(xq, wqh, bq, 0.125f, qf, nullptr);
  proj_gemm_kernel<true ><<<gGemm, 256, 0, stream>>>(xk, wkh, bk, 1.0f,   kf, nullptr);
  proj_gemm_kernel<true ><<<gGemm, 256, 0, stream>>>(xv, wvh, bv, 1.0f,   vf, nullptr);

  transpose_v_kernel<<<dim3(SEQ / 16, HDIM / 16, BATCH * HEADS), 256, 0, stream>>>(vf, vt);

  flash_attn_kernel<<<dim3(BATCH * HEADS, SEQ / 128), 256, 0, stream>>>(qf, kf, vt, rbias, attn16);

  proj_gemm_kernel<false><<<gGemm, 256, 0, stream>>>(attn16, woh, bo, 1.0f, nullptr, (float*)d_out);
}
